// BaseBiRNN_21492016349299
// MI455X (gfx1250) — compile-verified, matrix operand order checked by reading, not run on hardware
//
#include <hip/hip_runtime.h>

// ---------------------------------------------------------------------------
// Bidirectional LSTM for MI455X (gfx1250), bf16 WMMA with fp32 accumulation.
//   B=64, T=512, D=512, H=768, gates G=4H=3072, K = D+H = 1280
// All fp32->bf16 conversion is hoisted out of the recurrent GEMM loop:
//   - seq is pre-converted to bf16 once (prep kernel)
//   - h keeps a bf16 shadow copy written by the step epilogue
// The recurrent GEMM loop is software-pipelined one k-chunk deep so the four
// gate-weight loads of chunk kc+1 are in flight while chunk kc's WMMAs run.
// ---------------------------------------------------------------------------

#define Bc 64
#define Tc 512
#define Dc 512
#define Hc 768
#define Gc 3072          // 4*H
#define KCH 40           // K chunks of 32 (1280/32)
#define KCX 16           // chunks covering x_t (512/32)
#define NTG 48           // 16-wide N tiles per gate (768/16)
#define NTA 192          // total N tiles over all 4 gates (3072/16)

typedef __bf16 bf16_t;
typedef __attribute__((ext_vector_type(16))) __bf16 v16bf;
typedef __attribute__((ext_vector_type(8)))  __bf16 v8bf;
typedef __attribute__((ext_vector_type(4)))  __bf16 v4bf;
typedef __attribute__((ext_vector_type(8)))  float  v8f;

// float -> bf16, round-to-nearest-even, branchless (inputs are finite)
__device__ __forceinline__ bf16_t f2bf(float f) {
    unsigned u = __float_as_uint(f);
    u += 0x7fffu + ((u >> 16) & 1u);
    unsigned short h = (unsigned short)(u >> 16);
    bf16_t r;
    __builtin_memcpy(&r, &h, sizeof(r));
    return r;
}

__device__ __forceinline__ float sigm(float x) {
    return 1.0f / (1.0f + __expf(-x));
}

#define WMMA_BF16(A, Bx, C) \
    __builtin_amdgcn_wmma_f32_16x16x32_bf16(false, (A), false, (Bx), (short)0, (C), false, false)

// ---------------------------------------------------------------------------
// Pack W (fp32, [1280 x 3072] row-major) into per-lane bf16 WMMA B-fragments.
// Layout: wp[dir][n_tile(192)][k_chunk(40)][lane(32)][16 bf16]  (32B per lane)
// ---------------------------------------------------------------------------
__global__ void pack_weights_kernel(const float* __restrict__ Wfw,
                                    const float* __restrict__ Wbw,
                                    bf16_t* __restrict__ wp) {
    const int NE  = NTA * KCH * 32 * 16;            // 3,932,160 per direction
    int tid = blockIdx.x * 256 + threadIdx.x;
    if (tid >= NE) return;
    int dir = blockIdx.y;
    const float* W = dir ? Wbw : Wfw;

    int e    = tid & 15;
    int lane = (tid >> 4) & 31;
    int kcnt = tid >> 9;                            // nt*KCH + kc
    int kc   = kcnt % KCH;
    int nt   = kcnt / KCH;                          // 0..191
    int n    = nt * 16 + (lane & 15);
    int lo   = (lane & 16) ? 8 : 0;
    int kk   = (e < 8) ? (lo + e) : (16 + lo + (e - 8));
    int k    = kc * 32 + kk;

    wp[(size_t)dir * NE + tid] = f2bf(W[(size_t)k * Gc + n]);
}

// Convert the whole sequence [B,T,D] fp32 -> bf16 (4 elements per thread).
__global__ void cvt_seq_kernel(const float* __restrict__ s,
                               bf16_t* __restrict__ d) {
    size_t i = ((size_t)blockIdx.x * 256 + threadIdx.x) * 4;
    float4 v = *(const float4*)(s + i);
    v4bf r;
    r[0] = f2bf(v.x); r[1] = f2bf(v.y); r[2] = f2bf(v.z); r[3] = f2bf(v.w);
    *(v4bf*)(d + i) = r;
}

__global__ void zero_b32_kernel(unsigned* __restrict__ p, int n) {
    int i = blockIdx.x * 256 + threadIdx.x;
    if (i < n) p[i] = 0u;
}

// ---------------------------------------------------------------------------
// One recurrent time step, both directions (blockIdx.y = dir).
// grid: (24, 2) x 256 threads = 192 waves/dir.
// Wave wid -> m_tile = wid&3 (16 batch rows), nt = wid>>2 (16 H columns).
// Accumulates the i/j/f/o gate tiles (shared A fragment), then does the LSTM
// cell update + masked/reversed output write in its epilogue.
// ---------------------------------------------------------------------------
__global__ __launch_bounds__(256)
void lstm_step_kernel(const bf16_t* __restrict__ seqbf,
                      const int*    __restrict__ slen,
                      const float*  __restrict__ bias_fw,
                      const float*  __restrict__ bias_bw,
                      const bf16_t* __restrict__ wpack,
                      float*        __restrict__ cstate,
                      const float*  __restrict__ hprev,
                      float*        __restrict__ hnext,
                      const bf16_t* __restrict__ hbfprev,
                      bf16_t*       __restrict__ hbfnext,
                      float*        __restrict__ out,
                      int t) {
    const int lane = threadIdx.x & 31;
    const int wid  = blockIdx.x * (blockDim.x >> 5) + (threadIdx.x >> 5); // 0..191
    const int dir  = blockIdx.y;
    const int mt   = wid & 3;        // M tile (batch)
    const int nt   = wid >> 2;       // N tile within one gate (H columns)

    // ---- A-fragment addressing (row = batch, K = [x_t | h_prev]) ----------
    const int mrow = mt * 16 + (lane & 15);
    const int Lr   = slen[mrow];
    const int tpos = (dir == 0) ? t : ((t < Lr) ? (Lr - 1 - t) : t);
    const int lo   = (lane & 16) ? 8 : 0;

    // Lane-adjusted bf16 source pointers: per chunk the lane reads
    // src[lo .. +7] and src[16 + lo .. +7]  (2 x 16B loads).
    const bf16_t* xsrc = seqbf + ((size_t)mrow * Tc + tpos) * Dc + lo;
    const bf16_t* hsrc = hbfprev + ((size_t)dir * Bc + mrow) * Hc + lo;

    // Per-gate B-fragment base pointers (stride between k-chunks = 32 v16bf).
    const v16bf* wpd = (const v16bf*)wpack + (size_t)dir * (NTA * KCH * 32);
    const v16bf* bp0 = wpd + ((size_t)(0 * NTG + nt) * KCH) * 32 + lane;
    const v16bf* bp1 = wpd + ((size_t)(1 * NTG + nt) * KCH) * 32 + lane;
    const v16bf* bp2 = wpd + ((size_t)(2 * NTG + nt) * KCH) * 32 + lane;
    const v16bf* bp3 = wpd + ((size_t)(3 * NTG + nt) * KCH) * 32 + lane;

    auto load_a = [&](int kc) -> v16bf {
        const bf16_t* s = (kc < KCX) ? (xsrc + (size_t)kc * 32)
                                     : (hsrc + (size_t)(kc - KCX) * 32);
        v8bf a0 = *(const v8bf*)(s);
        v8bf a1 = *(const v8bf*)(s + 16);
        return __builtin_shufflevector(a0, a1,
            0, 1, 2, 3, 4, 5, 6, 7, 8, 9, 10, 11, 12, 13, 14, 15);
    };

    v8f acc0, acc1, acc2, acc3;
    const v8f vz = {0.f, 0.f, 0.f, 0.f, 0.f, 0.f, 0.f, 0.f};
    acc0 = vz; acc1 = vz; acc2 = vz; acc3 = vz;

    // ---- software-pipelined GEMM over K: prologue loads chunk 0 -----------
    v16bf a_c  = load_a(0);
    v16bf b0_c = bp0[0];
    v16bf b1_c = bp1[0];
    v16bf b2_c = bp2[0];
    v16bf b3_c = bp3[0];

#pragma unroll 2
    for (int kc = 0; kc < KCH - 1; ++kc) {
        // issue next chunk's loads before consuming the current fragments
        const size_t off = (size_t)(kc + 1) * 32;
        v16bf a_n  = load_a(kc + 1);
        v16bf b0_n = bp0[off];
        v16bf b1_n = bp1[off];
        v16bf b2_n = bp2[off];
        v16bf b3_n = bp3[off];

        acc0 = WMMA_BF16(a_c, b0_c, acc0);
        acc1 = WMMA_BF16(a_c, b1_c, acc1);
        acc2 = WMMA_BF16(a_c, b2_c, acc2);
        acc3 = WMMA_BF16(a_c, b3_c, acc3);

        a_c = a_n; b0_c = b0_n; b1_c = b1_n; b2_c = b2_n; b3_c = b3_n;
    }
    // epilogue chunk
    acc0 = WMMA_BF16(a_c, b0_c, acc0);
    acc1 = WMMA_BF16(a_c, b1_c, acc1);
    acc2 = WMMA_BF16(a_c, b2_c, acc2);
    acc3 = WMMA_BF16(a_c, b3_c, acc3);

    // ---- Epilogue: bias + nonlinearities + cell update + masked output ----
    const float* bias = (dir == 0) ? bias_fw : bias_bw;
    const int ncol = nt * 16 + (lane & 15);
    const float b_i = bias[ncol];
    const float b_j = bias[Hc + ncol];
    const float b_f = bias[2 * Hc + ncol] + 1.0f;   // TF forget bias
    const float b_o = bias[3 * Hc + ncol];
    const int mhi = (lane >> 4) & 1;

    float*        cst  = cstate  + (size_t)dir * Bc * Hc;
    const float*  hpb  = hprev   + (size_t)dir * Bc * Hc;
    float*        hnb  = hnext   + (size_t)dir * Bc * Hc;
    bf16_t*       hbfn = hbfnext + (size_t)dir * Bc * Hc;

#pragma unroll
    for (int r = 0; r < 8; ++r) {
        const int brow = mt * 16 + r + 8 * mhi;      // batch row of this element
        const int Lb   = slen[brow];
        const size_t sidx = (size_t)brow * Hc + ncol;

        const float cprev = cst[sidx];
        const float hpv   = hpb[sidx];

        const float gi = sigm (acc0[r] + b_i);
        const float gj = tanhf(acc1[r] + b_j);
        const float gf = sigm (acc2[r] + b_f);
        const float go = sigm (acc3[r] + b_o);

        const float cn = cprev * gf + gi * gj;
        const float hv = tanhf(cn) * go;

        const bool  valid = (t < Lb);
        const float hcar  = valid ? hv : hpv;        // carried hidden state
        if (valid) cst[sidx] = cn;                   // c frozen past seq_len
        hnb[sidx]  = hcar;                           // fp32 h for carry/state
        hbfn[sidx] = f2bf(hcar);                     // bf16 h for next matmul

        // fw writes pos t; bw scatters to L-1-t (valid) and zero-fills pos t
        // (invalid) so every (b, pos) of d_out is written exactly once.
        const int pos = (dir == 0) ? t : (valid ? (Lb - 1 - t) : t);
        const float ov = valid ? hv : 0.0f;
        out[((size_t)brow * Tc + pos) * (2 * Hc) + dir * Hc + ncol] = ov;
    }
}

__global__ void write_state_kernel(const float* __restrict__ hfinal,
                                   float* __restrict__ out) {
    int tid = blockIdx.x * 256 + threadIdx.x;
    const int total = 2 * Bc * Hc;
    if (tid >= total) return;
    int dir = tid / (Bc * Hc);
    int rem = tid % (Bc * Hc);
    int b   = rem / Hc;
    int j   = rem % Hc;
    out[(size_t)Bc * Tc * (2 * Hc) + (size_t)b * (2 * Hc) + dir * Hc + j] = hfinal[tid];
}

// ---------------------------------------------------------------------------
extern "C" void kernel_launch(void* const* d_in, const int* in_sizes, int n_in,
                              void* d_out, int out_size, void* d_ws, size_t ws_size,
                              hipStream_t stream) {
    const float* seq  = (const float*)d_in[0];   // [B,T,D] f32
    const int*   slen = (const int*)  d_in[1];   // [B] i32
    const float* Wfw  = (const float*)d_in[2];   // [1280,3072] f32
    const float* bfw  = (const float*)d_in[3];   // [3072] f32
    const float* Wbw  = (const float*)d_in[4];
    const float* bbw  = (const float*)d_in[5];
    float* out = (float*)d_out;

    // ---- workspace layout (all regions 256B-aligned by construction) ------
    const int    NE     = NTA * KCH * 32 * 16;               // packed elems / dir
    const size_t SEQN   = (size_t)Bc * Tc * Dc;              // 16,777,216
    char* ws = (char*)d_ws;
    bf16_t* wpack = (bf16_t*)ws;                ws += (size_t)2 * NE * sizeof(bf16_t);  // 15.7 MB
    bf16_t* seqbf = (bf16_t*)ws;                ws += SEQN * sizeof(bf16_t);            // 33.6 MB
    float*  cst   = (float*)ws;                 ws += (size_t)2 * Bc * Hc * 4;
    float*  hb0   = (float*)ws;                 ws += (size_t)2 * Bc * Hc * 4;
    float*  hb1   = (float*)ws;                 ws += (size_t)2 * Bc * Hc * 4;
    bf16_t* hbf0  = (bf16_t*)ws;                ws += (size_t)2 * Bc * Hc * 2;
    bf16_t* hbf1  = (bf16_t*)ws;                ws += (size_t)2 * Bc * Hc * 2;
    // states are contiguous: zero them all as dwords
    const int state_dw = (int)(((char*)ws - (char*)cst) / 4);

    // 1) Prep: pack weights, convert seq to bf16, zero c/h state.
    pack_weights_kernel<<<dim3((NE + 255) / 256, 2), 256, 0, stream>>>(Wfw, Wbw, wpack);
    cvt_seq_kernel<<<dim3((unsigned)(SEQN / 4 / 256)), 256, 0, stream>>>(seq, seqbf);
    zero_b32_kernel<<<dim3((state_dw + 255) / 256), 256, 0, stream>>>((unsigned*)cst, state_dw);

    // 2) Recurrent scan: one fused gates+cell kernel per step, both directions.
    for (int t = 0; t < Tc; ++t) {
        const float*  hp  = (t & 1) ? hb1  : hb0;
        float*        hn  = (t & 1) ? hb0  : hb1;
        const bf16_t* hbp = (t & 1) ? hbf1 : hbf0;
        bf16_t*       hbn = (t & 1) ? hbf0 : hbf1;
        lstm_step_kernel<<<dim3(24, 2), 256, 0, stream>>>(
            seqbf, slen, bfw, bbw, wpack, cst, hp, hn, hbp, hbn, out, t);
    }

    // 3) Final hidden state (T even -> last write landed in hb0).
    write_state_kernel<<<dim3((2 * Bc * Hc + 255) / 256), 256, 0, stream>>>(hb0, out);
}
